// HebbianBlock_1348619731084
// MI455X (gfx1250) — matrix operand order repaired and checked by reading, not compiled, not run on hardware
//
#include <hip/hip_runtime.h>

typedef __bf16 bf16_t;
typedef __attribute__((ext_vector_type(16))) __bf16 v16bf;
typedef __attribute__((ext_vector_type(8)))  __bf16 v8bf;
typedef __attribute__((ext_vector_type(8)))  float  v8f;

#define Bn 4
#define Tn 8192
#define Dn 1024
#define Cn 64
#define nCn (Tn / Cn)
#define NBT ((size_t)Bn * Tn * Dn)
#define LOG2E 1.44269504088896340736f

__device__ __forceinline__ float fexp2(float x) { return __builtin_amdgcn_exp2f(x); }
__device__ __forceinline__ float flog2(float x) { return __builtin_amdgcn_logf(x); }

// ---------------------------------------------------------------------------
// WMMA helpers (CDNA5 wave32, v_wmma_f32_16x16x32_bf16)
// ---------------------------------------------------------------------------
__device__ __forceinline__ v8f wmma_bf16(v16bf a, v16bf b, v8f c) {
  return __builtin_amdgcn_wmma_f32_16x16x32_bf16(false, a, false, b, (short)0, c,
                                                 false, false);
}

// A fragment 16x32 (MxK), memory row-major mem[m*ld + k].
// Per-lane: m = lane&15, half = lane>>4; element j -> K = (j>>3)*16 + half*8 + (j&7)
// => two contiguous 8-element (16B) loads.
__device__ __forceinline__ v16bf ldA(const bf16_t* p, int ld, int lane) {
  int m = lane & 15, h = lane >> 4;
  const v8bf* q0 = (const v8bf*)(p + (size_t)m * ld + h * 8);
  const v8bf* q1 = (const v8bf*)(p + (size_t)m * ld + 16 + h * 8);
  v8bf a = *q0, b = *q1;
  v16bf r;
#pragma unroll
  for (int j = 0; j < 8; ++j) { r[j] = a[j]; r[j + 8] = b[j]; }
  return r;
}

// B fragment 32x16 (KxN) where memory holds B transposed: mem[n*ld + k].
// Per-lane: n = lane&15, half = lane>>4; element j -> K = (j>>3)*16 + half*8 + (j&7)
// => two contiguous 8-element (16B) loads.
__device__ __forceinline__ v16bf ldBt(const bf16_t* p, int ld, int lane) {
  int n = lane & 15, h = lane >> 4;
  const v8bf* q0 = (const v8bf*)(p + (size_t)n * ld + h * 8);
  const v8bf* q1 = (const v8bf*)(p + (size_t)n * ld + 16 + h * 8);
  v8bf a = *q0, b = *q1;
  v16bf r;
#pragma unroll
  for (int j = 0; j < 8; ++j) { r[j] = a[j]; r[j + 8] = b[j]; }
  return r;
}

// Same as ldBt but source is f32 (the LDS-resident W state) with on-the-fly cvt.
__device__ __forceinline__ v16bf ldBt_f32(const float* p, int ld, int lane) {
  int n = lane & 15, h = lane >> 4;
  const float* q = p + (size_t)n * ld;
  v16bf r;
#pragma unroll
  for (int j = 0; j < 16; ++j) r[j] = (bf16_t)q[(j >> 3) * 16 + h * 8 + (j & 7)];
  return r;
}

// ---------------------------------------------------------------------------
// Prep: rk = bf16(out); wk = bf16(shift-right-1 of out)
// ---------------------------------------------------------------------------
__global__ __launch_bounds__(256) void prep_cast(const float* __restrict__ out,
                                                 bf16_t* __restrict__ rk,
                                                 bf16_t* __restrict__ wk) {
  size_t idx = (size_t)blockIdx.x * 256 + threadIdx.x;
  if (idx >= NBT) return;
  int td = (int)(idx % (size_t)(Tn * Dn));
  int t = td / Dn;
  float x = out[idx];
  rk[idx] = (bf16_t)x;
  float xw = (t > 0) ? out[idx - Dn] : 0.f;
  wk[idx] = (bf16_t)xw;
}

// wkT[b][d][t] = wk[b][t][d]  (LDS-tiled transpose)
__global__ __launch_bounds__(256) void transpose_wk(const float* __restrict__ out,
                                                    bf16_t* __restrict__ wkT) {
  __shared__ bf16_t tile[32][33];
  int d0 = blockIdx.x * 32, t0 = blockIdx.y * 32, b = blockIdx.z;
  int tx = threadIdx.x & 31, ty = threadIdx.x >> 5;
  for (int i = ty; i < 32; i += 8) {
    int t = t0 + i;
    float x = (t > 0) ? out[((size_t)b * Tn + (t - 1)) * Dn + d0 + tx] : 0.f;
    tile[i][tx] = (bf16_t)x;
  }
  __syncthreads();
  for (int i = ty; i < 32; i += 8)
    wkT[((size_t)b * Dn + d0 + i) * Tn + t0 + tx] = tile[tx][i];
}

// Swizzled W^T for B-fragments: Ws[(e>>5)*32*D + d*32 + (e&31)] = W[d][e]
__global__ __launch_bounds__(256) void prep_w(const float* __restrict__ W,
                                              bf16_t* __restrict__ Ws) {
  int idx = blockIdx.x * 256 + threadIdx.x;
  int d = idx >> 10, e = idx & 1023;
  Ws[(size_t)(e >> 5) * 32 * Dn + (size_t)d * 32 + (e & 31)] = (bf16_t)W[idx];
}

// ---------------------------------------------------------------------------
// bf16 WMMA GEMM: C[M=32768, N=1024] = A @ Bsw  (K = 1024)
// Workgroup block 128x64; 8 waves in a 4x2 grid, each wave owns a 2x2 group of
// 16x16 tiles -> 4 WMMAs per 4 fragment loads per K-step (2x arithmetic
// intensity vs 2x1 strips).
// MODE 0: C -> bf16 buffer.  MODE 1: Cf = base + exp(log_alpha) * C (f32).
// ---------------------------------------------------------------------------
template <int MODE>
__global__ __launch_bounds__(256) void gemm64(const bf16_t* __restrict__ A,
                                              const bf16_t* __restrict__ Bs,
                                              bf16_t* __restrict__ Cbf,
                                              const float* __restrict__ base,
                                              float* __restrict__ Cf,
                                              const float* __restrict__ p_log_alpha) {
  int nb = blockIdx.x;   // N/64
  int mb = blockIdx.y;   // M/128
  int wave = threadIdx.x >> 5, lane = threadIdx.x & 31;
  int row0 = mb * 128 + (wave & 3) * 32;   // two m-tiles: row0, row0+16
  int col0 = nb * 64 + (wave >> 2) * 32;   // two n-tiles: col0, col0+16
  v8f acc00 = {}, acc01 = {}, acc10 = {}, acc11 = {};
  for (int k = 0; k < Dn; k += 32) {
    v16bf b0 = ldBt(Bs + (size_t)k * Dn + (size_t)col0 * 32, 32, lane);
    v16bf b1 = ldBt(Bs + (size_t)k * Dn + (size_t)(col0 + 16) * 32, 32, lane);
    v16bf a0 = ldA(A + (size_t)row0 * Dn + k, Dn, lane);
    v16bf a1 = ldA(A + (size_t)(row0 + 16) * Dn + k, Dn, lane);
    acc00 = wmma_bf16(a0, b0, acc00);
    acc01 = wmma_bf16(a0, b1, acc01);
    acc10 = wmma_bf16(a1, b0, acc10);
    acc11 = wmma_bf16(a1, b1, acc11);
  }
  int n = lane & 15, h = lane >> 4;
  if (MODE == 0) {
#pragma unroll
    for (int r = 0; r < 8; ++r) {
      int m0 = row0 + r + 8 * h;
      Cbf[(size_t)m0 * Dn + col0 + n]        = (bf16_t)acc00[r];
      Cbf[(size_t)m0 * Dn + col0 + 16 + n]   = (bf16_t)acc01[r];
      Cbf[(size_t)(m0 + 16) * Dn + col0 + n]      = (bf16_t)acc10[r];
      Cbf[(size_t)(m0 + 16) * Dn + col0 + 16 + n] = (bf16_t)acc11[r];
    }
  } else {
    float alpha = fexp2((*p_log_alpha) * LOG2E);
#pragma unroll
    for (int r = 0; r < 8; ++r) {
      int m0 = row0 + r + 8 * h;
      size_t i00 = (size_t)m0 * Dn + col0 + n;
      size_t i01 = i00 + 16;
      size_t i10 = (size_t)(m0 + 16) * Dn + col0 + n;
      size_t i11 = i10 + 16;
      Cf[i00] = base[i00] + alpha * acc00[r];
      Cf[i01] = base[i01] + alpha * acc01[r];
      Cf[i10] = base[i10] + alpha * acc10[r];
      Cf[i11] = base[i11] + alpha * acc11[r];
    }
  }
}

// ---------------------------------------------------------------------------
// Persistent chunk-scan kernel. 128 workgroups = 4 batches x 32 row-tiles of W.
// Each workgroup keeps its 32x1024 f32 slice of the Hebbian state W in LDS for
// all 128 chunks; inter-chunk dependency is purely intra-workgroup.
// ---------------------------------------------------------------------------
__global__ __launch_bounds__(256) void scan_kernel(
    const bf16_t* __restrict__ rk, const bf16_t* __restrict__ wk,
    const bf16_t* __restrict__ wkT, const bf16_t* __restrict__ vv,
    bf16_t* __restrict__ reads, const float* __restrict__ p_decay) {
  __shared__ __align__(16) float  Wt[32][1028];  // padded: banks spread
  __shared__ __align__(16) bf16_t vT[32][72];    // v^T tile (raw)
  __shared__ __align__(16) bf16_t sS[64][72];    // (S .* M) bf16

  int wg = blockIdx.x;
  int b = wg >> 5;
  int d0 = (wg & 31) * 32;
  int tid = threadIdx.x, wave = tid >> 5, lane = tid & 31;
  int n = lane & 15, h = lane >> 4;

  float gamma = 1.f / (1.f + fexp2(-(*p_decay) * LOG2E));
  float lg2 = flog2(gamma);
  float gCc = fexp2(lg2 * (float)Cn);

  for (int i = tid; i < 32 * 1028; i += 256) ((float*)Wt)[i] = 0.f;
  __syncthreads();

  const bf16_t* rk_b  = rk  + (size_t)b * Tn * Dn;
  const bf16_t* wk_b  = wk  + (size_t)b * Tn * Dn;
  const bf16_t* wkT_b = wkT + (size_t)b * Dn * Tn;
  const bf16_t* v_b   = vv  + (size_t)b * Tn * Dn;
  bf16_t*       rd_b  = reads + (size_t)b * Tn * Dn;

  // per-phase wave partitions (all uniform across the wave -> EXEC all ones)
  int niS = wave & 3, miS0 = (wave >> 2) * 2;     // S: 64x64 -> 2 tiles/wave
  int mi = wave >> 1, ni = wave & 1;              // inter/intra: 64x32 -> 1 tile/wave
  int umi = wave & 1; int un0 = (wave >> 1) * 16; // update: 32x1024 -> 16 tiles/wave

  for (int ck = 0; ck < nCn; ++ck) {
    int t0 = ck * Cn;

    // ---- stage v^T tile: vT[d][c] = v[t0+c][d0+d]
    for (int i = tid; i < Cn * 32; i += 256) {
      int c = i >> 5, d = i & 31;
      vT[d][c] = v_b[(size_t)(t0 + c) * Dn + d0 + d];
    }

    // ---- S = r @ w^T (full 64x64, redundant per d-tile, cheap), mask with M
    v8f s0 = {}, s1 = {};
    for (int kk = 0; kk < Dn; kk += 32) {
      v16bf bfrag = ldBt(wk_b + (size_t)(t0 + niS * 16) * Dn + kk, Dn, lane);
      v16bf a0 = ldA(rk_b + (size_t)(t0 + miS0 * 16) * Dn + kk, Dn, lane);
      v16bf a1 = ldA(rk_b + (size_t)(t0 + miS0 * 16 + 16) * Dn + kk, Dn, lane);
      s0 = wmma_bf16(a0, bfrag, s0);
      s1 = wmma_bf16(a1, bfrag, s1);
    }
    {
      int cc = niS * 16 + n;
#pragma unroll
      for (int r = 0; r < 8; ++r) {
        int c0v = miS0 * 16 + r + 8 * h;
        sS[c0v][cc] = (bf16_t)((c0v > cc)
                         ? s0[r] * fexp2(lg2 * (float)(c0v - 1 - cc)) : 0.f);
        int c1v = c0v + 16;
        sS[c1v][cc] = (bf16_t)((c1v > cc)
                         ? s1[r] * fexp2(lg2 * (float)(c1v - 1 - cc)) : 0.f);
      }
    }
    __syncthreads();

    // ---- inter = (r @ Wt^T) * gamma^c, then += intra = (S.*M) @ v
    v8f acc = {};
    for (int kk = 0; kk < Dn; kk += 32) {
      v16bf bfrag = ldBt_f32(&Wt[ni * 16][kk], 1028, lane);
      v16bf a0 = ldA(rk_b + (size_t)(t0 + mi * 16) * Dn + kk, Dn, lane);
      acc = wmma_bf16(a0, bfrag, acc);
    }
#pragma unroll
    for (int r = 0; r < 8; ++r)
      acc[r] *= fexp2(lg2 * (float)(mi * 16 + r + 8 * h));
    for (int kk = 0; kk < Cn; kk += 32) {
      v16bf bfrag = ldBt(&vT[ni * 16][kk], 72, lane);
      v16bf a0 = ldA(&sS[mi * 16][kk], 72, lane);
      acc = wmma_bf16(a0, bfrag, acc);
    }
    {
      int dcol = d0 + ni * 16 + n;
#pragma unroll
      for (int r = 0; r < 8; ++r)
        rd_b[(size_t)(t0 + mi * 16 + r + 8 * h) * Dn + dcol] = (bf16_t)acc[r];
    }
    __syncthreads();  // all inter-reads of Wt done before update writes

    // ---- update: Wt = gamma^C * Wt + (v .* gw)^T @ w
    v16bf ua0, ua1;
    {
      const bf16_t* p = &vT[umi * 16 + n][0];
#pragma unroll
      for (int j = 0; j < 16; ++j) {
        int K0 = (j >> 3) * 16 + h * 8 + (j & 7);
        ua0[j] = (bf16_t)((float)p[K0] * fexp2(lg2 * (float)(Cn - 1 - K0)));
        int K1 = 32 + K0;
        ua1[j] = (bf16_t)((float)p[K1] * fexp2(lg2 * (float)(Cn - 1 - K1)));
      }
    }
    for (int nt = 0; nt < 16; ++nt) {
      int e0 = (un0 + nt) * 16;
      v8f wacc;
#pragma unroll
      for (int r = 0; r < 8; ++r)
        wacc[r] = gCc * Wt[umi * 16 + r + 8 * h][e0 + n];
      v16bf b0 = ldBt(wkT_b + (size_t)e0 * Tn + t0, Tn, lane);
      v16bf b1 = ldBt(wkT_b + (size_t)e0 * Tn + t0 + 32, Tn, lane);
      wacc = wmma_bf16(ua0, b0, wacc);
      wacc = wmma_bf16(ua1, b1, wacc);
#pragma unroll
      for (int r = 0; r < 8; ++r)
        Wt[umi * 16 + r + 8 * h][e0 + n] = wacc[r];
    }
    __syncthreads();  // Wt/vT/sS stable before next chunk
  }
}

// ---------------------------------------------------------------------------
// Host-side launch (graph-capture safe: kernels only, all on `stream`)
// ---------------------------------------------------------------------------
extern "C" void kernel_launch(void* const* d_in, const int* in_sizes, int n_in,
                              void* d_out, int out_size, void* d_ws, size_t ws_size,
                              hipStream_t stream) {
  (void)in_sizes; (void)n_in; (void)out_size; (void)ws_size;
  const float* out   = (const float*)d_in[0];
  const float* Ww    = (const float*)d_in[1];
  const float* Wr    = (const float*)d_in[2];
  const float* decay = (const float*)d_in[3];
  const float* logal = (const float*)d_in[4];
  float* dout = (float*)d_out;

  char* ws = (char*)d_ws;
  size_t off = 0;
  bf16_t* rk  = (bf16_t*)(ws + off); off += NBT * 2;            // bf16(out)
  bf16_t* wk  = (bf16_t*)(ws + off); off += NBT * 2;            // shifted keys
  bf16_t* wkT = (bf16_t*)(ws + off); off += NBT * 2;            // shifted keys^T
  bf16_t* vv  = (bf16_t*)(ws + off); off += NBT * 2;            // write projection
  bf16_t* rd  = (bf16_t*)(ws + off); off += NBT * 2;            // scan output
  bf16_t* WwS = (bf16_t*)(ws + off); off += (size_t)Dn * Dn * 2;
  bf16_t* WrS = (bf16_t*)(ws + off); off += (size_t)Dn * Dn * 2;

  prep_cast<<<(unsigned)(NBT / 256), 256, 0, stream>>>(out, rk, wk);
  transpose_wk<<<dim3(Dn / 32, Tn / 32, Bn), 256, 0, stream>>>(out, wkT);
  prep_w<<<(Dn * Dn) / 256, 256, 0, stream>>>(Ww, WwS);
  prep_w<<<(Dn * Dn) / 256, 256, 0, stream>>>(Wr, WrS);

  // v = out @ W_write^T  (bf16 in/out, f32 accumulate)
  gemm64<0><<<dim3(Dn / 64, (Bn * Tn) / 128), 256, 0, stream>>>(
      rk, WwS, vv, nullptr, nullptr, nullptr);

  // chunked Hebbian scan (single persistent launch, W state in LDS)
  scan_kernel<<<Bn * (Dn / 32), 256, 0, stream>>>(rk, wk, wkT, vv, rd, decay);

  // d_out = out + exp(log_alpha) * (reads @ W_read^T)
  gemm64<1><<<dim3(Dn / 64, (Bn * Tn) / 128), 256, 0, stream>>>(
      rd, WrS, nullptr, out, dout, logal);
}
